// AttentionHead_47047071760964
// MI455X (gfx1250) — matrix operand order repaired and checked by reading, not compiled, not run on hardware
//
#include <hip/hip_runtime.h>
#include <hip/hip_bf16.h>

#define DIM_EMB 1024
#define DIM_K   128
#define BATCH   4
#define SEQ     4096
#define MTOT    (BATCH * SEQ)

typedef __attribute__((ext_vector_type(16))) __bf16 v16bf;
typedef __attribute__((ext_vector_type(8)))  float  v8f;

union BF16x16 {
    v16bf v;
    uint4 q[2];
    unsigned short h[16];
};

// float -> bf16 bits, round-to-nearest-even
__device__ __forceinline__ unsigned short f2bf_u(float f) {
    unsigned u = __builtin_bit_cast(unsigned, f);
    u += 0x7fffu + ((u >> 16) & 1u);
    return (unsigned short)(u >> 16);
}

// Load a 16-element bf16 fragment from two contiguous 16-byte runs.
__device__ __forceinline__ v16bf load_bf16x16(const unsigned short* p0,
                                              const unsigned short* p1) {
    BF16x16 u;
    u.q[0] = *reinterpret_cast<const uint4*>(p0);
    u.q[1] = *reinterpret_cast<const uint4*>(p1);
    return u.v;
}

// ---------------------------------------------------------------------------
// Kernel 0: transpose Wq/Wk/Wv ([1024,128] f32, row-major) -> wT[o][n][k] bf16
// ---------------------------------------------------------------------------
__global__ void __launch_bounds__(256) build_wT(
    const float* __restrict__ Wq, const float* __restrict__ Wk,
    const float* __restrict__ Wv, unsigned short* __restrict__ wT) {
    int i = blockIdx.x * 256 + threadIdx.x;
    if (i >= 3 * DIM_K * DIM_EMB) return;
    const int o = i >> 17;                      // 128*1024 = 2^17
    const int r = i & (DIM_K * DIM_EMB - 1);
    const int n = r >> 10;
    const int k = r & (DIM_EMB - 1);
    const float* W = (o == 0) ? Wq : ((o == 1) ? Wk : Wv);
    wT[i] = f2bf_u(W[(size_t)k * DIM_K + n]);
}

// ---------------------------------------------------------------------------
// Kernel 1: q/k/v projection. One wave computes a 16x128 output strip of one
// of {q,k,v}. sched_barrier(0) pins all fragment loads above the WMMA chain
// so the waits retire in stages (loadcnt N..0) instead of serializing.
// ---------------------------------------------------------------------------
__global__ void __launch_bounds__(32) qkv_proj(
    const float* __restrict__ x, const unsigned short* __restrict__ wT,
    unsigned short* __restrict__ qo, unsigned short* __restrict__ ko,
    unsigned short* __restrict__ vT) {
    const int lane = threadIdx.x & 31;
    const int nl = lane & 15, hi = lane >> 4;
    const int m0 = blockIdx.x * 16;
    const int o  = blockIdx.y;
    const unsigned short* wTo = wT + (size_t)o * DIM_K * DIM_EMB;
    const float* xrow = x + (size_t)(m0 + nl) * DIM_EMB;

    v8f c[8] = {};
    #pragma unroll 2
    for (int k0 = 0; k0 < DIM_EMB; k0 += 32) {
        // ---- load region: A source (f32) + all 8 B fragments (bf16) ----
        const float* xa = xrow + k0 + 8 * hi;
        const float4 fA = *reinterpret_cast<const float4*>(xa + 0);
        const float4 fB = *reinterpret_cast<const float4*>(xa + 4);
        const float4 fC = *reinterpret_cast<const float4*>(xa + 16);
        const float4 fD = *reinterpret_cast<const float4*>(xa + 20);

        v16bf bfr[8];
        #pragma unroll
        for (int t = 0; t < 8; ++t) {
            const unsigned short* wr =
                wTo + (size_t)(t * 16 + nl) * DIM_EMB + k0 + 8 * hi;
            bfr[t] = load_bf16x16(wr, wr + 16);
        }
        // Keep all loads hoisted above the compute region.
        __builtin_amdgcn_sched_barrier(0);

        // ---- compute region: convert A, then the WMMA chain ----
        const float fs[16] = {fA.x, fA.y, fA.z, fA.w, fB.x, fB.y, fB.z, fB.w,
                              fC.x, fC.y, fC.z, fC.w, fD.x, fD.y, fD.z, fD.w};
        BF16x16 a;
        #pragma unroll
        for (int i = 0; i < 16; ++i) a.h[i] = f2bf_u(fs[i]);

        #pragma unroll
        for (int t = 0; t < 8; ++t)
            c[t] = __builtin_amdgcn_wmma_f32_16x16x32_bf16(
                false, a.v, false, bfr[t], (short)0, c[t], false, false);
    }

    // C layout: VGPR v, lanes 0-15 -> row m0+v ; lanes 16-31 -> row m0+8+v
    if (o < 2) {
        unsigned short* out = (o == 0) ? qo : ko;
        #pragma unroll
        for (int t = 0; t < 8; ++t)
            #pragma unroll
            for (int v = 0; v < 8; ++v)
                out[(size_t)(m0 + v + 8 * hi) * DIM_K + t * 16 + nl] =
                    f2bf_u(c[t][v]);
    } else {
        const int bb = m0 >> 12, s = m0 & (SEQ - 1);
        #pragma unroll
        for (int t = 0; t < 8; ++t)
            #pragma unroll
            for (int v = 0; v < 8; ++v)
                vT[((size_t)bb * DIM_K + t * 16 + nl) * SEQ + s + v + 8 * hi] =
                    f2bf_u(c[t][v]);
    }
}

// ---------------------------------------------------------------------------
// Kernel 2: causal flash-attention. One wave owns 16 query rows x dv=128.
// Regions per 32-key block:
//   [K loads] | [S-WMMAs + V loads + prefetch] | [softmax + LDS] [PV-WMMAs]
// so V-load latency hides under softmax and next-iteration K loads can
// interleave with PV WMMAs.
// ---------------------------------------------------------------------------
__global__ void __launch_bounds__(128) attn_fwd(
    const unsigned short* __restrict__ qb, const unsigned short* __restrict__ kb,
    const unsigned short* __restrict__ vT, float* __restrict__ out) {
    __shared__ __align__(16) unsigned short p_lds[4][16][32];
    const int wave = threadIdx.x >> 5;
    const int lane = threadIdx.x & 31;
    const int nl = lane & 15, hi = lane >> 4;
    const int b  = blockIdx.y;
    const int q0 = (blockIdx.x * 4 + wave) * 16;
    const unsigned short* qB = qb + (size_t)b * SEQ * DIM_K;
    const unsigned short* kB = kb + (size_t)b * SEQ * DIM_K;
    const unsigned short* vB = vT + (size_t)b * DIM_K * SEQ;
    const float SCALE  = 0.03125f;   // 1/sqrt(DIM_EMB) per reference
    const float NEGBIG = -3.0e38f;

    // Q fragments held in registers: 16 rows x 128 dims = 4 A-fragments
    v16bf qa[4];
    const unsigned short* qr = qB + (size_t)(q0 + nl) * DIM_K;
    #pragma unroll
    for (int kk = 0; kk < 4; ++kk)
        qa[kk] = load_bf16x16(qr + kk * 32 + 8 * hi, qr + kk * 32 + 16 + 8 * hi);

    v8f c[8] = {};
    float mrow[8], lrow[8];
    #pragma unroll
    for (int v = 0; v < 8; ++v) { mrow[v] = NEGBIG; lrow[v] = 0.0f; }

    for (int k0 = 0; k0 <= q0 + 15; k0 += 32) {
        // ---- load region: all 8 K fragments for this block ----
        v16bf kf0[4], kf1[4];
        #pragma unroll
        for (int kk = 0; kk < 4; ++kk) {
            const unsigned short* kr0 =
                kB + (size_t)(k0 + nl) * DIM_K + kk * 32 + 8 * hi;
            const unsigned short* kr1 =
                kB + (size_t)(k0 + 16 + nl) * DIM_K + kk * 32 + 8 * hi;
            kf0[kk] = load_bf16x16(kr0, kr0 + 16);
            kf1[kk] = load_bf16x16(kr1, kr1 + 16);
        }
        __builtin_amdgcn_sched_barrier(0);

        // ---- S-WMMAs; V loads interleave here and retire under softmax ----
        v8f s0 = {}, s1 = {};
        #pragma unroll
        for (int kk = 0; kk < 4; ++kk) {
            s0 = __builtin_amdgcn_wmma_f32_16x16x32_bf16(
                false, qa[kk], false, kf0[kk], (short)0, s0, false, false);
            s1 = __builtin_amdgcn_wmma_f32_16x16x32_bf16(
                false, qa[kk], false, kf1[kk], (short)0, s1, false, false);
        }
        v16bf vf[8];
        #pragma unroll
        for (int t = 0; t < 8; ++t) {
            const unsigned short* vr =
                vB + (size_t)(t * 16 + nl) * SEQ + k0 + 8 * hi;
            vf[t] = load_bf16x16(vr, vr + 16);
        }
        // speculative prefetch of next K block (global_prefetch_b8)
        __builtin_prefetch(kB + (size_t)(k0 + 32 + nl) * DIM_K, 0, 0);
        __builtin_amdgcn_sched_barrier(0);

        // ---- online softmax in C layout ----
        #pragma unroll
        for (int v = 0; v < 8; ++v) {
            const int qg = q0 + v + 8 * hi;                   // my query row
            float a0 = (k0 + nl      <= qg) ? s0[v] * SCALE : NEGBIG;
            float a1 = (k0 + 16 + nl <= qg) ? s1[v] * SCALE : NEGBIG;
            // row max over the 16 lanes holding this row's keys
            float mp = fmaxf(a0, a1);
            mp = fmaxf(mp, __shfl_xor(mp, 1, 32));
            mp = fmaxf(mp, __shfl_xor(mp, 2, 32));
            mp = fmaxf(mp, __shfl_xor(mp, 4, 32));
            mp = fmaxf(mp, __shfl_xor(mp, 8, 32));
            const float mnew  = fmaxf(mrow[v], mp);
            const float alpha = __expf(mrow[v] - mnew);
            const float p0 = __expf(a0 - mnew);
            const float p1 = __expf(a1 - mnew);
            float ps = p0 + p1;
            ps += __shfl_xor(ps, 1, 32);
            ps += __shfl_xor(ps, 2, 32);
            ps += __shfl_xor(ps, 4, 32);
            ps += __shfl_xor(ps, 8, 32);
            lrow[v] = lrow[v] * alpha + ps;
            mrow[v] = mnew;
            // C-layout -> plain [m][k] tile in LDS (bf16)
            p_lds[wave][v + 8 * hi][nl]      = f2bf_u(p0);
            p_lds[wave][v + 8 * hi][16 + nl] = f2bf_u(p1);
            // rescale running output
            #pragma unroll
            for (int t = 0; t < 8; ++t) c[t][v] *= alpha;
        }
        // in-wave LDS store->load ordering
        asm volatile("s_wait_dscnt 0" ::: "memory");
        // P as A-fragment: lane nl = row m, k runs {8*hi..}, {16+8*hi..}
        v16bf pa = load_bf16x16(&p_lds[wave][nl][8 * hi],
                                &p_lds[wave][nl][16 + 8 * hi]);
        #pragma unroll
        for (int t = 0; t < 8; ++t)
            c[t] = __builtin_amdgcn_wmma_f32_16x16x32_bf16(
                false, pa, false, vf[t], (short)0, c[t], false, false);
    }

    float* outB = out + (size_t)b * SEQ * DIM_K;
    #pragma unroll
    for (int t = 0; t < 8; ++t)
        #pragma unroll
        for (int v = 0; v < 8; ++v)
            outB[(size_t)(q0 + v + 8 * hi) * DIM_K + t * 16 + nl] =
                c[t][v] / lrow[v];
}

// ---------------------------------------------------------------------------
extern "C" void kernel_launch(void* const* d_in, const int* in_sizes, int n_in,
                              void* d_out, int out_size, void* d_ws,
                              size_t ws_size, hipStream_t stream) {
    (void)in_sizes; (void)n_in; (void)out_size; (void)ws_size;
    const float* x  = (const float*)d_in[0];
    const float* Wq = (const float*)d_in[1];
    const float* Wk = (const float*)d_in[2];
    const float* Wv = (const float*)d_in[3];
    float* out = (float*)d_out;

    // workspace layout (bf16): q[16384*128] k[16384*128] vT[4*128*4096] wT[3*128*1024]
    unsigned short* qbf = (unsigned short*)d_ws;
    unsigned short* kbf = qbf + (size_t)MTOT * DIM_K;
    unsigned short* vT  = kbf + (size_t)MTOT * DIM_K;
    unsigned short* wT  = vT  + (size_t)MTOT * DIM_K;

    build_wT<<<(3 * DIM_K * DIM_EMB + 255) / 256, 256, 0, stream>>>(Wq, Wk, Wv, wT);
    qkv_proj<<<dim3(MTOT / 16, 3), 32, 0, stream>>>(x, wT, qbf, kbf, vT);
    attn_fwd<<<dim3(SEQ / 16 / 4, BATCH), 128, 0, stream>>>(qbf, kbf, vT, out);
}
